// SP_UNet_90950227460827
// MI455X (gfx1250) — matrix operand order repaired and checked by reading, not compiled, not run on hardware
//
#include <hip/hip_runtime.h>

typedef __attribute__((ext_vector_type(16))) __bf16 v16bf;
typedef __attribute__((ext_vector_type(8)))  float  v8f;
typedef __attribute__((vector_size(16)))     int    v4i_vs;
typedef unsigned short u16;
typedef unsigned char  u8;

#ifndef __has_builtin
#define __has_builtin(x) 0
#endif
#if __has_builtin(__builtin_amdgcn_global_load_async_to_lds_b128)
#define HAVE_ASYNC_LDS 1
#else
#define HAVE_ASYNC_LDS 0
#endif

union ABFrag { v16bf v; uint4 u[2]; };
union AccU   { v8f v; float f[8]; };

__device__ __forceinline__ u16 f2bf(float f) {
  unsigned int u = __float_as_uint(f);
  unsigned int r = (u + 0x7FFFu + ((u >> 16) & 1u)) >> 16;
  return (u16)r;
}

// ---------------------------------------------------------------------------
// Implicit-GEMM conv, 128 threads = 4 waves per block.
// Wave wv computes M-tile (blockIdx.x*4+wv): 16 output pixels x 16 out chans;
// all 4 waves share the N-tile.  The K loop is flattened over (tap, 64-chunk)
// and software-pipelined: the 64x16 bf16 weight chunk for step j+1 is staged
// into LDS with global_load_async_to_lds_b128 while step j feeds two
// v_wmma_f32_16x16x32_bf16 ops from the other LDS buffer (16-bit LDS reads).
// One s_wait_asynccnt + one barrier per 2 WMMAs.
// All dims are powers of two -> shift/mask indexing only.
// A-frag (16-bit 16x32): lane<16 -> K {0..7,16..23}, lane>=16 -> {8..15,24..31},
// row M = lane&15.  B-frag: lane&15 = N, element s -> same K as A.  D: lane&15 = N, vgpr r ->
// M = r + 8*(lane>=16).
// ---------------------------------------------------------------------------
__global__ __launch_bounds__(128)
void conv_wmma_kernel(const u16* __restrict__ in, const u16* __restrict__ w,
                      const float* __restrict__ bias,
                      float* __restrict__ outF, u16* __restrict__ outB,
                      const u8* __restrict__ mask,
                      int Hin, int Win, int logWin, int Cin, int logCin,
                      int Wout, int logWout, int Cout, int logCout,
                      int KH, int KW, int stride, int pad) {
  __shared__ __align__(16) u16 bs[2][1024];   // double-buffered 64(K) x 16(N)
  const int tid = threadIdx.x;
  const int lane = tid & 31;
  const int wv = tid >> 5;
  const int mrow = lane & 15;
  const int half = lane >> 4;
  const int tileM = blockIdx.x * 4 + wv;
  const int n0 = blockIdx.y * 16;
  const int p = tileM * 16 + mrow;
  const int oy = p >> logWout, ox = p & (Wout - 1);

  // cooperative stage indices: thread -> (row 0..63, col 0/8), 16B each
  const int sk = tid >> 1;
  const int sq = (tid & 1) * 8;

  const int logCpt = logCin - 6;          // 64-chunks per tap (power of two)
  const int cptM1 = (1 << logCpt) - 1;
  const int nch = (KH * KW) << logCpt;

  auto stage = [&](int j) {
    const int tap = j >> logCpt;
    const int c0 = (j & cptM1) << 6;
    const u16* g = w + ((size_t)((tap << logCin) + c0 + sk) << logCout) + n0 + sq;
    u16* d = &bs[j & 1][sk * 16 + sq];
#if HAVE_ASYNC_LDS
    __builtin_amdgcn_global_load_async_to_lds_b128(
        (__attribute__((address_space(1))) v4i_vs*)g,
        (__attribute__((address_space(3))) v4i_vs*)d, 0, 0);
#else
    *(uint4*)d = *(const uint4*)g;
#endif
  };

  stage(0);
  v8f acc = {};
  for (int j = 0; j < nch; ++j) {
#if HAVE_ASYNC_LDS
    asm volatile("s_wait_asynccnt 0x0" ::: "memory");   // chunk j landed
#endif
    __syncthreads();                                    // visible to all waves
    if (j + 1 < nch) stage(j + 1);                      // overlap with compute

    const int tap = j >> logCpt;
    const int c0 = (j & cptM1) << 6;
    const int kh = (KW == 3) ? ((tap * 11) >> 5) : tap; // exact for tap<9
    const int kw = tap - kh * KW;
    const int iy = oy * stride - pad + kh;
    const int ix = ox * stride - pad + kw;
    const bool inb = (iy >= 0) & (iy < Hin) & (ix >= 0) & (ix < Win);

    ABFrag a0, a1, b0, b1;
    if (inb) {
      const u16* src = in + ((size_t)((iy << logWin) + ix) << logCin) + c0;
      __builtin_prefetch((const void*)(src + 64), 0, 0);  // global_prefetch_b8
      a0.u[0] = *(const uint4*)(src + half * 8);
      a0.u[1] = *(const uint4*)(src + 16 + half * 8);
      a1.u[0] = *(const uint4*)(src + 32 + half * 8);
      a1.u[1] = *(const uint4*)(src + 48 + half * 8);
    } else {
      a0.u[0].x = a0.u[0].y = a0.u[0].z = a0.u[0].w = 0u;
      a0.u[1] = a0.u[0]; a1.u[0] = a0.u[0]; a1.u[1] = a0.u[0];
    }
    const u16* lb = bs[j & 1];
    for (int s = 0; s < 16; ++s) {
      const int kk = (s < 8) ? (half * 8 + s) : (16 + half * 8 + (s - 8));
      b0.v[s] = __builtin_bit_cast(__bf16, lb[kk * 16 + mrow]);
      b1.v[s] = __builtin_bit_cast(__bf16, lb[(32 + kk) * 16 + mrow]);
    }
    acc = __builtin_amdgcn_wmma_f32_16x16x32_bf16(false, a0.v, false, b0.v,
                                                  (short)0, acc, false, false);
    acc = __builtin_amdgcn_wmma_f32_16x16x32_bf16(false, a1.v, false, b1.v,
                                                  (short)0, acc, false, false);
  }

  AccU res; res.v = acc;
  const int n = lane & 15;
  const float bval = bias ? bias[n0 + n] : 0.0f;
  for (int r = 0; r < 8; ++r) {
    const int m = half * 8 + r;
    const int q = tileM * 16 + m;
    float v = res.f[r] + bval;
    if (!mask[q]) v = 0.0f;
    const size_t o = ((size_t)q << logCout) + n0 + n;
    if (outF) outF[o] = v;
    else      outB[o] = f2bf(v);
  }
}

// Transposed conv 2x2 stride 2 VALID, no bias; blockIdx.z = tap (dy*2+dx).
// out[2y+dy][2x+dx][co] = sum_ci in[y][x][ci] * wu[dy][dx][ci][co]
__global__ __launch_bounds__(128)
void convt2_wmma_kernel(const u16* __restrict__ in, const u16* __restrict__ wu,
                        u16* __restrict__ out, const u8* __restrict__ mask,
                        int Win, int logWin, int C, int logC) {
  const int tid = threadIdx.x;
  const int lane = tid & 31;
  const int wv = tid >> 5;
  const int mrow = lane & 15;
  const int half = lane >> 4;
  const int tile = blockIdx.x * 4 + wv;
  const int n0 = blockIdx.y * 16;
  const int tap = blockIdx.z;
  const int dy = tap >> 1, dx = tap & 1;
  const int p = tile * 16 + mrow;

  v8f acc = {};
  const u16* src = in + ((size_t)p << logC);
  const u16* wt = wu + (((size_t)tap << logC) << logC) + n0;
  for (int c0 = 0; c0 < C; c0 += 32) {
    ABFrag a, b;
    a.u[0] = *(const uint4*)(src + c0 + half * 8);
    a.u[1] = *(const uint4*)(src + c0 + 16 + half * 8);
    for (int s = 0; s < 16; ++s) {
      const int kk = (s < 8) ? (half * 8 + s) : (16 + half * 8 + (s - 8));
      b.v[s] = __builtin_bit_cast(__bf16, wt[((size_t)(c0 + kk) << logC) + mrow]);
    }
    acc = __builtin_amdgcn_wmma_f32_16x16x32_bf16(false, a.v, false, b.v,
                                                  (short)0, acc, false, false);
  }
  AccU res; res.v = acc;
  const int n = lane & 15;
  for (int r = 0; r < 8; ++r) {
    const int m = half * 8 + r;
    const int q = tile * 16 + m;
    const int yy = 2 * (q >> logWin) + dy;
    const int xx = 2 * (q & (Win - 1)) + dx;
    const size_t op = ((size_t)yy << (logWin + 1)) + xx;
    float v = res.f[r];
    if (!mask[op]) v = 0.0f;
    out[(op << logC) + n0 + n] = f2bf(v);
  }
}

// ---------------------------------------------------------------------------
// BN: deterministic two-stage masked stats (activations masked to zero, so
// plain sums == masked sums).  part rows: [sum(C) | sumsq(C) | count(1)]
// ---------------------------------------------------------------------------
__global__ void bn_stats_partial(const float* __restrict__ x,
                                 const u8* __restrict__ mask,
                                 float* __restrict__ part,
                                 int Npix, int C, int chunk) {
  const int c = threadIdx.x;            // blockDim.x == C (<=256)
  const int b = blockIdx.x;
  int p0 = b * chunk;
  int p1 = p0 + chunk; if (p1 > Npix) p1 = Npix;
  float s = 0.f, s2 = 0.f;
  for (int p = p0; p < p1; ++p) {
    float v = x[(size_t)p * C + c];
    s += v; s2 += v * v;
  }
  float* row = part + (size_t)b * (2 * C + 1);
  row[c] = s; row[C + c] = s2;
  if (c == 0) {
    float cnt = 0.f;
    for (int p = p0; p < p1; ++p) cnt += mask[p] ? 1.f : 0.f;
    row[2 * C] = cnt;
  }
}

__global__ void bn_stats_final(const float* __restrict__ part,
                               float* __restrict__ stats, int nb, int C) {
  const int c = threadIdx.x;
  float s = 0.f, s2 = 0.f, cnt = 0.f;
  for (int b = 0; b < nb; ++b) {
    const float* row = part + (size_t)b * (2 * C + 1);
    s += row[c]; s2 += row[C + c];
    if (c == 0) cnt += row[2 * C];
  }
  stats[c] = s; stats[C + c] = s2;
  if (c == 0) stats[2 * C] = cnt;
}

__global__ void bn_apply_kernel(const float* __restrict__ x,
                                const u8* __restrict__ mask,
                                const float* __restrict__ stats,
                                const float* __restrict__ gamma,
                                const float* __restrict__ beta,
                                u16* __restrict__ y, int Npix, int C, int logC) {
  size_t i = (size_t)blockIdx.x * blockDim.x + threadIdx.x;
  if (i >= ((size_t)Npix << logC)) return;
  const int c = (int)(i & (C - 1));
  const size_t p = i >> logC;
  float cnt = stats[2 * C]; if (cnt < 1.f) cnt = 1.f;
  const float mean = stats[c] / cnt;
  const float var = stats[C + c] / cnt - mean * mean;
  float v = (x[i] - mean) * rsqrtf(var + 1e-5f) * gamma[c] + beta[c];
  v = (v > 0.f) ? v : 0.01f * v;          // LeakyReLU(0.01)
  if (!mask[p]) v = 0.f;
  y[i] = f2bf(v);
}

// ---------------------------------------------------------------------------
// Elementwise utilities (all shift/mask indexed)
// ---------------------------------------------------------------------------
__global__ void cvt_in_kernel(const float* __restrict__ x,
                              const u8* __restrict__ m,
                              u16* __restrict__ y, int Npix, int logC) {
  size_t i = (size_t)blockIdx.x * blockDim.x + threadIdx.x;
  if (i >= ((size_t)Npix << logC)) return;
  y[i] = m[i >> logC] ? f2bf(x[i]) : (u16)0;
}

__global__ void cvt_w_kernel(const float* __restrict__ w, u16* __restrict__ o, int n) {
  int i = blockIdx.x * blockDim.x + threadIdx.x;
  if (i < n) o[i] = f2bf(w[i]);
}

__global__ void poolmask_kernel(const u8* __restrict__ m, u8* __restrict__ o,
                                int Wo, int logWo) {
  int i = blockIdx.x * blockDim.x + threadIdx.x;
  if (i >= Wo * Wo) return;
  const int y = i >> logWo, x = i & (Wo - 1);
  const int base = ((2 * y) << (logWo + 1)) + 2 * x;
  const int W = 2 * Wo;
  u8 v = m[base] | m[base + 1] | m[base + W] | m[base + W + 1];
  o[i] = v ? 1 : 0;
}

__global__ void ormask_kernel(const u8* a, const u8* b, u8* o, int n) {
  int i = blockIdx.x * blockDim.x + threadIdx.x;
  if (i < n) o[i] = (a[i] | b[i]) ? 1 : 0;
}

__global__ void concat_kernel(const u16* __restrict__ a, const u8* __restrict__ ma,
                              const u16* __restrict__ b, const u8* __restrict__ mb,
                              u16* __restrict__ o, int Npix, int C, int logC) {
  size_t i = (size_t)blockIdx.x * blockDim.x + threadIdx.x;
  if (i >= ((size_t)Npix << (logC + 1))) return;
  const size_t p = i >> (logC + 1);
  const int c = (int)(i & (2 * C - 1));
  u16 v;
  if (c < C) v = ma[p] ? a[(p << logC) + c] : (u16)0;
  else       v = mb[p] ? b[(p << logC) + (c - C)] : (u16)0;
  o[i] = v;
}

// ---------------------------------------------------------------------------
// Host orchestration
// ---------------------------------------------------------------------------
namespace {
struct Arena {
  char* base; size_t off; size_t cap;
  void* get(size_t bytes) {
    size_t a = (off + 255) & ~(size_t)255;
    off = a + bytes;
    return base + a;
  }
};
struct EncLayer { u16* w; const float *b, *g, *be; };
struct DecParams { u16* w1; const float* b1; u16* wu; u16* w2; const float* b2;
                   u16* w3; const float* b3; u16* w4; const float* b4; };
static int ilog2i(int v) { int l = 0; while ((1 << l) < v) ++l; return l; }
} // namespace

extern "C" void kernel_launch(void* const* d_in, const int* in_sizes, int n_in,
                              void* d_out, int out_size, void* d_ws, size_t ws_size,
                              hipStream_t stream) {
  (void)in_sizes; (void)n_in; (void)out_size;

  const float* pc0 = (const float*)d_in[0];
  const u8*    m0  = (const u8*)d_in[1];
  const float* pc1 = (const float*)d_in[2];
  const u8*    m1  = (const u8*)d_in[3];
  int pi = 4;
  auto P = [&]() { return (const float*)d_in[pi++]; };

  Arena A{(char*)d_ws, 0, ws_size};

  float* stats = (float*)A.get(513 * sizeof(float));
  float* part  = (float*)A.get((size_t)256 * 513 * sizeof(float));

  auto cvtW = [&](const float* w, size_t n) -> u16* {
    u16* o = (u16*)A.get(n * 2);
    cvt_w_kernel<<<dim3((unsigned)((n + 255) / 256)), 256, 0, stream>>>(w, o, (int)n);
    return o;
  };

  // ---- params (setup_inputs insertion order, depth-first) ----
  EncLayer enc1[5], enc2[6], enc3[6];
  for (int i = 0; i < 5; ++i) { enc1[i].w = cvtW(P(), (size_t)9 * 64 * 64);
    enc1[i].b = P(); enc1[i].g = P(); enc1[i].be = P(); }
  for (int i = 0; i < 6; ++i) { enc2[i].w = cvtW(P(), (size_t)9 * (i == 0 ? 64 : 128) * 128);
    enc2[i].b = P(); enc2[i].g = P(); enc2[i].be = P(); }
  for (int i = 0; i < 6; ++i) { enc3[i].w = cvtW(P(), (size_t)9 * (i == 0 ? 128 : 256) * 256);
    enc3[i].b = P(); enc3[i].g = P(); enc3[i].be = P(); }

  DecParams dec[3];
  const int skipC[3] = {512, 256, 128};
  const int latC[3]  = {256, 128, 128};
  const int outC[3]  = {256, 128, 64};
  for (int k = 0; k < 3; ++k) {
    dec[k].w1 = cvtW(P(), (size_t)skipC[k] * latC[k]); dec[k].b1 = P();
    dec[k].wu = cvtW(P(), (size_t)4 * latC[k] * latC[k]);
    dec[k].w2 = cvtW(P(), (size_t)latC[k] * latC[k]);   dec[k].b2 = P();
    dec[k].w3 = cvtW(P(), (size_t)9 * 2 * latC[k] * outC[k]); dec[k].b3 = P();
    dec[k].w4 = cvtW(P(), (size_t)9 * outC[k] * outC[k]);     dec[k].b4 = P();
  }
  u16* d4w = cvtW(P(), (size_t)9 * 64 * 64);
  const float* d4b = P();

  // ---- launch helpers ----
  auto conv = [&](const u16* in, const u16* w, const float* bias,
                  float* outF, u16* outB, const u8* mask,
                  int Hin, int Cin, int Hout, int Cout, int KH, int s, int pad) {
    dim3 g((unsigned)(Hout * Hout / 64), (unsigned)(Cout / 16));
    conv_wmma_kernel<<<g, 128, 0, stream>>>(
        in, w, bias, outF, outB, mask,
        Hin, Hin, ilog2i(Hin), Cin, ilog2i(Cin),
        Hout, ilog2i(Hout), Cout, ilog2i(Cout), KH, KH, s, pad);
  };
  auto bnConv = [&](const u16* in, int Hin, int Cin, const EncLayer& L,
                    int Cout, int s, const u8* mask, float* tmpF, u16* outAct) {
    const int Hout = Hin / s;
    const int Npix = Hout * Hout;
    conv(in, L.w, L.b, tmpF, nullptr, mask, Hin, Cin, Hout, Cout, 3, s, 1);
    const int chunk = 64;
    const int nb = (Npix + chunk - 1) / chunk;   // <= 256
    bn_stats_partial<<<dim3((unsigned)nb), (unsigned)Cout, 0, stream>>>(
        tmpF, mask, part, Npix, Cout, chunk);
    bn_stats_final<<<1, (unsigned)Cout, 0, stream>>>(part, stats, nb, Cout);
    bn_apply_kernel<<<dim3((unsigned)(((size_t)Npix * Cout + 255) / 256)), 256, 0,
                      stream>>>(tmpF, mask, stats, L.g, L.be, outAct, Npix, Cout,
                                ilog2i(Cout));
  };
  auto pool = [&](const u8* m, u8* o, int Ho) {
    poolmask_kernel<<<dim3((unsigned)((Ho * Ho + 255) / 256)), 256, 0, stream>>>(
        m, o, Ho, ilog2i(Ho));
  };
  auto orm = [&](const u8* a, const u8* b, int n) -> u8* {
    u8* o = (u8*)A.get((size_t)n);
    ormask_kernel<<<dim3((unsigned)((n + 255) / 256)), 256, 0, stream>>>(a, b, o, n);
    return o;
  };
  auto cat = [&](const u16* a, const u8* ma, const u16* b, const u8* mb,
                 int Npix, int C) -> u16* {
    u16* o = (u16*)A.get((size_t)Npix * 2 * C * 2);
    concat_kernel<<<dim3((unsigned)(((size_t)Npix * 2 * C + 255) / 256)), 256, 0,
                    stream>>>(a, ma, b, mb, o, Npix, C, ilog2i(C));
    return o;
  };

  // ---- encoder (shared weights, run twice) ----
  struct EncOut { u16 *x, *F, *L, *R; u8 *mF, *mL, *mR; };
  auto encode = [&](const float* feat, const u8* mask) -> EncOut {
    EncOut E;
    E.x = (u16*)A.get((size_t)256 * 256 * 64 * 2);
    cvt_in_kernel<<<dim3((unsigned)(((size_t)256 * 256 * 64 + 255) / 256)), 256, 0,
                    stream>>>(feat, mask, E.x, 256 * 256, 6);
    E.mF = (u8*)A.get(128 * 128); pool(mask, E.mF, 128);
    E.mL = (u8*)A.get(64 * 64);   pool(E.mF, E.mL, 64);
    E.mR = (u8*)A.get(32 * 32);   pool(E.mL, E.mR, 32);
    float* tmp = (float*)A.get((size_t)128 * 128 * 64 * sizeof(float));

    // enc1: 5 layers @128^2, 64ch (first conv stride 2 from 256^2)
    u16* a1 = (u16*)A.get((size_t)128 * 128 * 64 * 2);
    u16* b1 = (u16*)A.get((size_t)128 * 128 * 64 * 2);
    bnConv(E.x, 256, 64, enc1[0], 64, 2, E.mF, tmp, a1);
    bnConv(a1, 128, 64, enc1[1], 64, 1, E.mF, tmp, b1);
    bnConv(b1, 128, 64, enc1[2], 64, 1, E.mF, tmp, a1);
    bnConv(a1, 128, 64, enc1[3], 64, 1, E.mF, tmp, b1);
    bnConv(b1, 128, 64, enc1[4], 64, 1, E.mF, tmp, a1);
    E.F = a1;

    // enc2: 6 layers @64^2, 128ch
    u16* a2 = (u16*)A.get((size_t)64 * 64 * 128 * 2);
    u16* b2 = (u16*)A.get((size_t)64 * 64 * 128 * 2);
    bnConv(E.F, 128, 64, enc2[0], 128, 2, E.mL, tmp, a2);
    bnConv(a2, 64, 128, enc2[1], 128, 1, E.mL, tmp, b2);
    bnConv(b2, 64, 128, enc2[2], 128, 1, E.mL, tmp, a2);
    bnConv(a2, 64, 128, enc2[3], 128, 1, E.mL, tmp, b2);
    bnConv(b2, 64, 128, enc2[4], 128, 1, E.mL, tmp, a2);
    bnConv(a2, 64, 128, enc2[5], 128, 1, E.mL, tmp, b2);
    E.L = b2;

    // enc3: 6 layers @32^2, 256ch
    u16* a3 = (u16*)A.get((size_t)32 * 32 * 256 * 2);
    u16* b3 = (u16*)A.get((size_t)32 * 32 * 256 * 2);
    bnConv(E.L, 64, 128, enc3[0], 256, 2, E.mR, tmp, a3);
    bnConv(a3, 32, 256, enc3[1], 256, 1, E.mR, tmp, b3);
    bnConv(b3, 32, 256, enc3[2], 256, 1, E.mR, tmp, a3);
    bnConv(a3, 32, 256, enc3[3], 256, 1, E.mR, tmp, b3);
    bnConv(b3, 32, 256, enc3[4], 256, 1, E.mR, tmp, a3);
    bnConv(a3, 32, 256, enc3[5], 256, 1, E.mR, tmp, b3);
    E.R = b3;
    return E;
  };

  EncOut E0 = encode(pc0, m0);
  EncOut E1 = encode(pc1, m1);

  // ---- skip concatenations (union masks) ----
  u8* mRu = orm(E0.mR, E1.mR, 32 * 32);
  u8* mLu = orm(E0.mL, E1.mL, 64 * 64);
  u8* mFu = orm(E0.mF, E1.mF, 128 * 128);
  u8* mBu = orm(m0, m1, 256 * 256);
  u16* Rs = cat(E0.R, E0.mR, E1.R, E1.mR, 32 * 32, 256);   // 512 ch
  u16* Ls = cat(E0.L, E0.mL, E1.L, E1.mL, 64 * 64, 128);   // 256 ch
  u16* Fs = cat(E0.F, E0.mF, E1.F, E1.mF, 128 * 128, 64);  // 128 ch
  u16* Bs = cat(E0.x, m0, E1.x, m1, 256 * 256, 64);        // 128 ch

  // ---- decoder upsample blocks ----
  auto upsample = [&](const u16* a, const u8* ma, int Ha, int Ca,
                      const u16* b, const u8* mb, int Hb, int Cb,
                      const DecParams& D, int lat, int Co) -> u16* {
    u16* fa = (u16*)A.get((size_t)Ha * Ha * lat * 2);
    conv(a, D.w1, D.b1, nullptr, fa, ma, Ha, Ca, Ha, lat, 1, 1, 0);
    u16* faU = (u16*)A.get((size_t)Hb * Hb * lat * 2);
    {
      dim3 g((unsigned)(Ha * Ha / 64), (unsigned)(lat / 16), 4);
      convt2_wmma_kernel<<<g, 128, 0, stream>>>(fa, D.wu, faU, mb,
                                                Ha, ilog2i(Ha), lat, ilog2i(lat));
    }
    u16* fb = (u16*)A.get((size_t)Hb * Hb * lat * 2);
    conv(b, D.w2, D.b2, nullptr, fb, mb, Hb, Cb, Hb, lat, 1, 1, 0);
    u16* c = cat(faU, mb, fb, mb, Hb * Hb, lat);            // 2*lat ch
    u16* c1 = (u16*)A.get((size_t)Hb * Hb * Co * 2);
    conv(c, D.w3, D.b3, nullptr, c1, mb, Hb, 2 * lat, Hb, Co, 3, 1, 1);
    u16* c2 = (u16*)A.get((size_t)Hb * Hb * Co * 2);
    conv(c1, D.w4, D.b4, nullptr, c2, mb, Hb, Co, Hb, Co, 3, 1, 1);
    return c2;
  };

  u16* S = upsample(Rs, mRu, 32, 512, Ls, mLu, 64, 256, dec[0], 256, 256);
  u16* T = upsample(S,  mLu, 64, 256, Fs, mFu, 128, 128, dec[1], 128, 128);
  u16* U = upsample(T,  mFu, 128, 128, Bs, mBu, 256, 128, dec[2], 128, 64);

  // ---- final stride-2 conv -> masked f32 straight to d_out ----
  u8* mV = (u8*)A.get(128 * 128);
  pool(mBu, mV, 128);
  conv(U, d4w, d4b, (float*)d_out, nullptr, mV, 256, 64, 128, 64, 3, 2, 1);
}
